// CLAHE_21294447854022
// MI455X (gfx1250) — matrix operand order repaired and checked by reading, not compiled, not run on hardware
//
#include <hip/hip_runtime.h>

typedef __attribute__((ext_vector_type(16))) _Float16 v16h;
typedef __attribute__((ext_vector_type(8)))  float    v8f;
typedef __attribute__((ext_vector_type(4)))  float    f4;

#define B_      128
#define HW      512
#define G_      8
#define TS      64
#define NB      256
#define CLIPLIM 80u
#define NTILES  (B_ * G_ * G_)   // 8192 tiles

// ---------------------------------------------------------------------------
// Pass 1: gray conversion + per-tile 256-bin histogram (LDS atomics).
// One block per 64x64 tile; 256 threads; each thread does 4 iters of a
// 4-consecutive-pixel group => 3x float4 coalesced NT loads (read-once
// stream must not flush the 192MB L2), packed u8x4 gray store (RT: stays
// L2-resident for pass 3), ds_add_u32 histogram increments.
// ---------------------------------------------------------------------------
__global__ __launch_bounds__(256) void clahe_hist(
    const float* __restrict__ in, unsigned* __restrict__ ghist,
    unsigned* __restrict__ gray4)
{
#pragma clang fp contract(off)
    __shared__ unsigned hist[NB];
    const int t = threadIdx.x;
    hist[t] = 0u;
    __syncthreads();

    const int tile = blockIdx.x;
    const int b  = tile >> 6;
    const int ty = (tile >> 3) & 7;
    const int tx = tile & 7;
    const int r0 = t >> 4;                 // 0..15 (row within 16-row slab)
    const int cg = t & 15;                 // 0..15 (group of 4 cols)
    const int col = tx * TS + cg * 4;

    for (int it = 0; it < 4; ++it) {
        const int row = ty * TS + it * 16 + r0;
        const int pix = (b * HW + row) * HW + col;          // pixel index, %4==0
        const f4* p = (const f4*)(in + (long)pix * 3);
        f4 q0 = __builtin_nontemporal_load(p + 0);
        f4 q1 = __builtin_nontemporal_load(p + 1);
        f4 q2 = __builtin_nontemporal_load(p + 2);
        float c[12] = {q0.x,q0.y,q0.z,q0.w, q1.x,q1.y,q1.z,q1.w, q2.x,q2.y,q2.z,q2.w};
        unsigned pack = 0u;
#pragma unroll
        for (int j = 0; j < 4; ++j) {
            // reference: clip(x,0,255).astype(int32) -> truncation
            int bb = (int)fminf(fmaxf(c[j*3+0], 0.f), 255.f);
            int gg = (int)fminf(fmaxf(c[j*3+1], 0.f), 255.f);
            int rr = (int)fminf(fmaxf(c[j*3+2], 0.f), 255.f);
            float gf = 0.114f*(float)bb + 0.587f*(float)gg + 0.299f*(float)rr;
            int gi = (int)fminf(fmaxf(rintf(gf), 0.f), 255.f);   // RNE == jnp.round
            atomicAdd(&hist[gi], 1u);
            pack |= (unsigned)gi << (8*j);
        }
        gray4[pix >> 2] = pack;
    }
    __syncthreads();
    ghist[tile * NB + t] = hist[t];
}

// ---------------------------------------------------------------------------
// Pass 2: clip + excess + cumsum + LUT, one wave (32 threads) per tile so
// EXEC is all-1s for WMMA. 256-wide cumsum done as two v_wmma_f32_16x16x32_f16:
//   P   = Lstrict x H            (P[m][n] = sum of full rows before m, col n)
//   cdf = [P | H] x [J ; U]      (all-rows-before total + in-row prefix)
// Clipped hist <=80 and P <=1200 are integers => exact in f16; f32 accumulate.
// Fractional redistributed excess added analytically: +(idx+1)*excess/256.
// ---------------------------------------------------------------------------
__global__ __launch_bounds__(32) void clahe_lut(
    const unsigned* __restrict__ ghist, float* __restrict__ lut)
{
#pragma clang fp contract(off)
    __shared__ float shmin[NB];   // clipped histogram H (as float)
    __shared__ float shp[NB];     // P staging for cross-lane redistribution
    const int lane = threadIdx.x;
    const int tile = blockIdx.x;
    const unsigned* hh = ghist + tile * NB;

    int ex = 0;
#pragma unroll
    for (int e = 0; e < 8; ++e) {
        int k = lane * 8 + e;
        unsigned h  = hh[k];
        unsigned mh = h < CLIPLIM ? h : CLIPLIM;
        ex += (int)(h - mh);
        shmin[k] = (float)mh;
    }
#pragma unroll
    for (int o = 16; o > 0; o >>= 1) ex += __shfl_xor(ex, o, 32);
    __syncthreads();

    const int half = lane >> 4;
    const int m = lane & 15;      // A-fragment row (M = lane%16 per ISA layout)
    const int n = lane & 15;      // B/C/D-fragment col

    // A as packed here (16-bit 16x32): elem e -> K = e + 16*half (contiguous, not ISA table order)
    // B as packed here (16-bit 32x16): elem e -> K = e + 16*half, N = lane%16 (same k order as A)
    v16h a1, b1;
#pragma unroll
    for (int e = 0; e < 16; ++e) {
        int k = 16*half + e;
        a1[e] = (_Float16)((k < 16 && k < m) ? 1.0f : 0.0f);      // Lstrict, K-padded
        b1[e] = (half == 0) ? (_Float16)shmin[16*e + n] : (_Float16)0.0f; // H on top
    }
    v8f zero = {0.f,0.f,0.f,0.f,0.f,0.f,0.f,0.f};
    v8f P = __builtin_amdgcn_wmma_f32_16x16x32_f16(false, a1, false, b1,
                                                   (short)0, zero, false, false);
    // C/D layout: elem v -> row M = v + 8*half, col N = lane%16
#pragma unroll
    for (int v = 0; v < 8; ++v) shp[(v + 8*half)*16 + n] = P[v];
    __syncthreads();

    v16h a2, b2;
#pragma unroll
    for (int e = 0; e < 16; ++e) {
        int k = 16*half + e;
        float av = (k < 16) ? shp[m*16 + k] : shmin[16*m + (k - 16)];  // [P | H]
        a2[e] = (_Float16)av;
        b2[e] = (half == 0) ? (_Float16)1.0f                           // J (ones)
                            : (_Float16)((e <= n) ? 1.0f : 0.0f);      // U (incl.)
    }
    v8f cdf = __builtin_amdgcn_wmma_f32_16x16x32_f16(false, a2, false, b2,
                                                     (short)0, zero, false, false);

    const float exf = (float)ex * (1.0f / 256.0f);
    float* lo = lut + tile * NB;
#pragma unroll
    for (int v = 0; v < 8; ++v) {
        int idx = (v + 8*half)*16 + n;
        float c = cdf[v] + (float)(idx + 1) * exf;
        float l = rintf(c * (255.0f / 4096.0f));
        lo[idx] = fminf(fmaxf(l, 0.0f), 255.0f);
    }
}

// ---------------------------------------------------------------------------
// Pass 3: bilinear LUT apply. Thread = 4 consecutive pixels: one u32 gray
// load (hopefully L2-hit), 16 L2-resident LUT gathers, 3x float4 coalesced
// NT output stores (write-once stream, keep L2 for gray/LUT).
// ---------------------------------------------------------------------------
__global__ __launch_bounds__(256) void clahe_apply(
    const unsigned* __restrict__ gray4, const float* __restrict__ lut,
    float* __restrict__ out)
{
#pragma clang fp contract(off)
    const int g = blockIdx.x * 256 + threadIdx.x;
    const int p = g * 4;
    const int b   = p >> 18;
    const int rem = p & 262143;
    const int y   = rem >> 9;
    const int x0  = rem & 511;
    const unsigned gp = gray4[g];

    float yf = (float)y * (1.0f/64.0f) - 0.5f;
    float y1 = floorf(yf);
    float ya = yf - y1;
    int y1i = min(max((int)y1, 0), 7);
    int y2i = min(max((int)y1 + 1, 0), 7);
    const float* lb = lut + b * (64 * 256);
    const float* l1 = lb + y1i * (8 * 256);
    const float* l2 = lb + y2i * (8 * 256);

    float res[12];
#pragma unroll
    for (int j = 0; j < 4; ++j) {
        int x = x0 + j;
        float xf = (float)x * (1.0f/64.0f) - 0.5f;
        float x1 = floorf(xf);
        float xa = xf - x1;
        int x1i = min(max((int)x1, 0), 7);
        int x2i = min(max((int)x1 + 1, 0), 7);
        int v = (int)((gp >> (8*j)) & 255u);
        float r11 = l1[x1i*256 + v];
        float r12 = l1[x2i*256 + v];
        float r21 = l2[x1i*256 + v];
        float r22 = l2[x2i*256 + v];
        float r = (r11*(1.f-xa) + r12*xa)*(1.f-ya) + (r21*(1.f-xa) + r22*xa)*ya;
        float o = fminf(fmaxf(rintf(r), 0.f), 255.f);
        res[3*j+0] = o; res[3*j+1] = o; res[3*j+2] = o;
    }
    f4* op = (f4*)(out + (long)p * 3);
    f4 o0 = {res[0], res[1], res[2],  res[3]};
    f4 o1 = {res[4], res[5], res[6],  res[7]};
    f4 o2 = {res[8], res[9], res[10], res[11]};
    __builtin_nontemporal_store(o0, op + 0);
    __builtin_nontemporal_store(o1, op + 1);
    __builtin_nontemporal_store(o2, op + 2);
}

// ---------------------------------------------------------------------------
extern "C" void kernel_launch(void* const* d_in, const int* in_sizes, int n_in,
                              void* d_out, int out_size, void* d_ws, size_t ws_size,
                              hipStream_t stream)
{
    const float* in  = (const float*)d_in[0];
    float*       out = (float*)d_out;
    char*        ws  = (char*)d_ws;

    unsigned* ghist = (unsigned*)ws;                                   // 8 MB
    float*    lut   = (float*)(ws + (size_t)NTILES * NB * 4);          // 8 MB
    unsigned* gray4 = (unsigned*)(ws + (size_t)NTILES * NB * 8);       // 33.5 MB

    clahe_hist <<<NTILES, 256, 0, stream>>>(in, ghist, gray4);
    clahe_lut  <<<NTILES,  32, 0, stream>>>(ghist, lut);
    const int npix = B_ * HW * HW;                                     // 33,554,432
    clahe_apply<<<(npix / 4) / 256, 256, 0, stream>>>(gray4, lut, out);
}
